// Phyloformer_33947421508055
// MI455X (gfx1250) — matrix operand order repaired and checked by reading, not compiled, not run on hardware
//
#include <hip/hip_runtime.h>
#include <math.h>

// ---------------- problem constants (from reference) ----------------
#define NB   6
#define Hh   4
#define Cc   192
#define Bb   2
#define Ss   64
#define Ll   256
#define Ntot (Bb*Ss*Ll)   /* 32768 positions, layout (b,s,l,c) row-major */
#define DH   48           /* head dim = C/H */
#define FF   768          /* 4*C */

typedef unsigned short u16;
typedef __attribute__((ext_vector_type(16))) __bf16 v16bf;
typedef __attribute__((ext_vector_type(8)))  __bf16 v8bf;
typedef __attribute__((ext_vector_type(8)))  float  v8f;

__device__ __forceinline__ u16 f2bf(float f) {
  union { float f; unsigned u; } v; v.f = f;
  unsigned u = v.u;
  unsigned r = u + 0x7FFFu + ((u >> 16) & 1u);   // round-to-nearest-even
  return (u16)(r >> 16);
}

// A fragment: 16x32 bf16 (ISA layout). lane(0-15)=M; idx0-7 -> K=k0+hi*8+idx,
// idx8-15 -> K=k0+16+hi*8+(idx-8)
__device__ __forceinline__ v16bf load_frag_a(const u16* __restrict__ arow, int k0, int hi) {
  v8bf lo = *(const v8bf*)(arow + k0 + hi * 8);
  v8bf up = *(const v8bf*)(arow + k0 + 16 + hi * 8);
  v16bf f;
#pragma unroll
  for (int i = 0; i < 8; i++) { f[i] = lo[i]; f[i + 8] = up[i]; }
  return f;
}

// B fragment: 32x16 bf16. N = lane%16, idx0-7 -> K=k0+hi*8+idx, idx8-15 -> K=k0+16+hi*8+(idx-8) ; B[k][n] = W[n][k]
__device__ __forceinline__ v16bf load_frag_b(const u16* __restrict__ wrow, int k0, int hi) {
  v8bf b0 = *(const v8bf*)(wrow + k0 + hi * 8);
  v8bf b1 = *(const v8bf*)(wrow + k0 + 16 + hi * 8);
  v16bf f;
#pragma unroll
  for (int i = 0; i < 8; i++) { f[i] = b0[i]; f[i + 8] = b1[i]; }
  return f;
}

__device__ __forceinline__ void load_step(const u16* __restrict__ arow0, const u16* __restrict__ arow1,
                                          const u16* __restrict__ w0, const u16* __restrict__ w1,
                                          const u16* __restrict__ w2, const u16* __restrict__ w3,
                                          int k0, int hi, v16bf (&a)[2], v16bf (&b)[4]) {
  a[0] = load_frag_a(arow0, k0, hi);
  a[1] = load_frag_a(arow1, k0, hi);
  b[0] = load_frag_b(w0, k0, hi);
  b[1] = load_frag_b(w1, k0, hi);
  b[2] = load_frag_b(w2, k0, hi);
  b[3] = load_frag_b(w3, k0, hi);
}

__device__ __forceinline__ void wmma8(const v16bf (&a)[2], const v16bf (&b)[4], v8f (&acc)[2][4]) {
#pragma unroll
  for (int mt = 0; mt < 2; mt++)
#pragma unroll
    for (int nt = 0; nt < 4; nt++)
      acc[mt][nt] = __builtin_amdgcn_wmma_f32_16x16x32_bf16(
          false, a[mt], false, b[nt], (short)0, acc[mt][nt], false, false);
}

// ---------------- f32 -> bf16 bulk convert (weights) ----------------
__global__ void cvt_bf16_kernel(const float* __restrict__ in, u16* __restrict__ out, int n) {
  int i = blockIdx.x * 256 + threadIdx.x;
  if (i < n) out[i] = f2bf(in[i]);
}

// ---------------- embedding + projection + rotary -------------------
__global__ __launch_bounds__(192)
void embed_kernel(const int* __restrict__ tokens, const float* __restrict__ emb,
                  const float* __restrict__ pw, const float* __restrict__ pb,
                  float* __restrict__ resid) {
  int n = blockIdx.x, c = threadIdx.x;
  int l = n & (Ll - 1);
  int tok = tokens[n];
  const float* e = emb + (size_t)tok * 64;
  const float* w = pw + (size_t)c * 64;
  float acc = pb[c];
#pragma unroll 8
  for (int j = 0; j < 64; j++) acc += e[j] * w[j];
  __shared__ float xr[Cc];
  xr[c] = acc;
  __syncthreads();
  int j = c % 96;
  float inv = powf(10000.0f, -(float)(2 * j) / 192.0f);
  float f = (float)l * inv;
  float other = (c < 96) ? -xr[c + 96] : xr[c - 96];
  resid[(size_t)n * Cc + c] = acc * cosf(f) + other * sinf(f);
}

// ---------------- channel LayerNorm -> bf16 -------------------------
__global__ __launch_bounds__(192)
void ln_kernel(const float* __restrict__ x, const float* __restrict__ g,
               const float* __restrict__ b, u16* __restrict__ out) {
  int n = blockIdx.x, c = threadIdx.x;
  float v = x[(size_t)n * Cc + c];
  __shared__ float s1[Cc], s2[Cc];
  s1[c] = v; s2[c] = v * v;
  __syncthreads();
  if (c < 32) {
    float a = 0.f, q = 0.f;
    for (int j = c; j < Cc; j += 32) { a += s1[j]; q += s2[j]; }
    s1[c] = a; s2[c] = q;
  }
  __syncthreads();
  if (c == 0) {
    float a = 0.f, q = 0.f;
    for (int j = 0; j < 32; j++) { a += s1[j]; q += s2[j]; }
    float m = a / Cc;
    s1[0] = m; s2[0] = q / Cc - m * m;
  }
  __syncthreads();
  float m = s1[0], var = s2[0];
  out[(size_t)n * Cc + c] = f2bf((v - m) * rsqrtf(var + 1e-5f) * g[c] + b[c]);
}

// ---------------- WMMA bf16 GEMM, fused epilogues -------------------
// D(M x Ncols) = A(M x K, bf16) * W(Ncols x K, bf16)^T + bias
// One wave computes a 32(M) x 64(N) tile: 8 accumulators.
// K is a template constant (192 / 768): the ping-pong two-buffer pipeline
// fully unrolls -> no register rotation moves, load addresses fold into
// 24-bit immediates, loads for step s+1 hoist over the 8 WMMAs of step s.
// grid = (M/32, Ncols/64).
// mode: 0 = write f32         (v projection)
//       1 = resid[m*192+n] += (o-proj, FFN2)
//       2 = gelu -> bf16      (FFN1)
//       3 = (elu+1)/sqrt(dh)  (q)
//       4 = elu+1             (k)
template<int K>
__global__ __launch_bounds__(32)
void gemm_wmma_kernel(const u16* __restrict__ A, const u16* __restrict__ W,
                      const float* __restrict__ bias,
                      float* __restrict__ outF, u16* __restrict__ outB,
                      float* __restrict__ resid, int mode) {
  constexpr int S = K / 32;         // number of K-steps (6 or 24, both even)
  const int lane  = threadIdx.x;
  const int hi    = lane >> 4;
  const int lm    = lane & 15;
  const int mBase = blockIdx.x * 32;
  const int nBase = blockIdx.y * 64;
  const int Ncols = (int)gridDim.y * 64;

  const u16* arow0 = A + (size_t)(mBase + lm) * K;
  const u16* arow1 = A + (size_t)(mBase + 16 + lm) * K;
  const u16* wrow0 = W + (size_t)(nBase + 0 * 16 + lm) * K;
  const u16* wrow1 = W + (size_t)(nBase + 1 * 16 + lm) * K;
  const u16* wrow2 = W + (size_t)(nBase + 2 * 16 + lm) * K;
  const u16* wrow3 = W + (size_t)(nBase + 3 * 16 + lm) * K;

  v8f acc[2][4];
#pragma unroll
  for (int mt = 0; mt < 2; mt++)
#pragma unroll
    for (int nt = 0; nt < 4; nt++)
      acc[mt][nt] = (v8f){0.f,0.f,0.f,0.f,0.f,0.f,0.f,0.f};

  v16bf a0[2], b0[4], a1[2], b1[4];
  load_step(arow0, arow1, wrow0, wrow1, wrow2, wrow3, 0, hi, a0, b0);

#pragma unroll
  for (int p = 0; p < S / 2 - 1; p++) {
    load_step(arow0, arow1, wrow0, wrow1, wrow2, wrow3, (2 * p + 1) * 32, hi, a1, b1);
    wmma8(a0, b0, acc);
    load_step(arow0, arow1, wrow0, wrow1, wrow2, wrow3, (2 * p + 2) * 32, hi, a0, b0);
    wmma8(a1, b1, acc);
  }
  load_step(arow0, arow1, wrow0, wrow1, wrow2, wrow3, (S - 1) * 32, hi, a1, b1);
  wmma8(a0, b0, acc);
  wmma8(a1, b1, acc);

#pragma unroll
  for (int mt = 0; mt < 2; mt++) {
#pragma unroll
    for (int nt = 0; nt < 4; nt++) {
      int n = nBase + nt * 16 + lm;
      float bv = bias[n];
#pragma unroll
      for (int r = 0; r < 8; r++) {
        int m = mBase + mt * 16 + hi * 8 + r;  // C/D layout: M=(lane/16)*8+r, N=lane%16
        float val = acc[mt][nt][r] + bv;
        if (mode == 0) {
          outF[(size_t)m * Ncols + n] = val;
        } else if (mode == 1) {
          resid[(size_t)m * Cc + n] += val;
        } else if (mode == 2) {
          float ge = 0.5f * val * (1.0f + erff(val * 0.70710678118f));
          outB[(size_t)m * Ncols + n] = f2bf(ge);
        } else if (mode == 3) {
          float e = val > 0.f ? val : (expf(val) - 1.f);
          outF[(size_t)m * Ncols + n] = (e + 1.f) * 0.14433756729740643f; // 1/sqrt(48)
        } else {
          float e = val > 0.f ? val : (expf(val) - 1.f);
          outF[(size_t)m * Ncols + n] = e + 1.f;
        }
      }
    }
  }
}

// ---------------- kv = k^T v and ksum per (group, head) -------------
__global__ __launch_bounds__(256)
void kv_reduce_kernel(const float* __restrict__ kf, const float* __restrict__ vf,
                      float* __restrict__ kvbuf, int rowMode, int Ngrp) {
  int gh = blockIdx.x;
  int h = gh & 3;
  int g = gh >> 2;
  int base, stride;
  if (rowMode) { base = g * Ll; stride = 1; }
  else { int b = g >> 8; int l = g & 255; base = b * (Ss * Ll) + l; stride = Ll; }
  int tid = threadIdx.x;
  __shared__ float kk[DH], vv[DH];
  float acc[9];
#pragma unroll
  for (int j = 0; j < 9; j++) acc[j] = 0.f;
  float ks = 0.f;
  for (int n = 0; n < Ngrp; n++) {
    int pos = base + n * stride;
    if (tid < 96) {
      int d = (tid < 48) ? tid : tid - 48;
      if (tid < 48) kk[d] = kf[(size_t)pos * Cc + h * DH + d];
      else          vv[d] = vf[(size_t)pos * Cc + h * DH + d];
    }
    __syncthreads();
#pragma unroll
    for (int j = 0; j < 9; j++) {
      int p = tid + 256 * j;                 // p in [0, 2304)
      acc[j] += kk[p / DH] * vv[p % DH];
    }
    if (tid < DH) ks += kk[tid];
    __syncthreads();
  }
  float* slot = kvbuf + (size_t)gh * (DH * DH + DH);
#pragma unroll
  for (int j = 0; j < 9; j++) slot[tid + 256 * j] = acc[j];
  if (tid < DH) slot[DH * DH + tid] = ks;
}

// ---------------- o = (q @ kv) / (q . ksum + eps) -> bf16 -----------
__global__ __launch_bounds__(192)
void o_compute_kernel(const float* __restrict__ qf, const float* __restrict__ kvbuf,
                      u16* __restrict__ ob, int rowMode) {
  int n = blockIdx.x;
  int c = threadIdx.x;
  int h = c / DH, e = c % DH;
  int g;
  if (rowMode) g = n / Ll;
  else { int b = n / (Ss * Ll); int l = n % Ll; g = b * Ll + l; }
  const float* slot = kvbuf + (size_t)(g * Hh + h) * (DH * DH + DH);
  __shared__ float q[Cc];
  q[c] = qf[(size_t)n * Cc + c];
  __syncthreads();
  const float* qh = q + h * DH;
  float denom = 1e-6f;
#pragma unroll 8
  for (int d = 0; d < DH; d++) denom += qh[d] * slot[DH * DH + d];
  float o = 0.f;
#pragma unroll 8
  for (int d = 0; d < DH; d++) o += qh[d] * slot[d * DH + e];
  ob[(size_t)n * Cc + c] = f2bf(o / denom);
}

// ---------------- final head: sigmoid(softplus(pw . x)) -------------
__global__ __launch_bounds__(128)
void final_kernel(const float* __restrict__ resid, const float* __restrict__ pw,
                  const float* __restrict__ pb, float* __restrict__ out) {
  int n = blockIdx.x * 128 + threadIdx.x;
  const float* r = resid + (size_t)n * Cc;
  float o = pb[0];
#pragma unroll 8
  for (int c = 0; c < Cc; c++) o += r[c] * pw[c];
  float sp = (o > 20.f) ? o : log1pf(expf(o));
  out[n] = 1.f / (1.f + expf(-sp));
}

// ====================================================================
extern "C" void kernel_launch(void* const* d_in, const int* in_sizes, int n_in,
                              void* d_out, int out_size, void* d_ws, size_t ws_size,
                              hipStream_t stream) {
  (void)in_sizes; (void)n_in; (void)out_size; (void)ws_size;
  const int*   tokens = (const int*)  d_in[0];
  const float* emb    = (const float*)d_in[1];
  const float* proj_w = (const float*)d_in[2];
  const float* proj_b = (const float*)d_in[3];
  const float* ln_g   = (const float*)d_in[4];
  const float* ln_b   = (const float*)d_in[5];
  const float* wq     = (const float*)d_in[6];
  const float* wk     = (const float*)d_in[7];
  const float* wv     = (const float*)d_in[8];
  const float* wo     = (const float*)d_in[9];
  const float* bq     = (const float*)d_in[10];
  const float* bk     = (const float*)d_in[11];
  const float* bv     = (const float*)d_in[12];
  const float* bo     = (const float*)d_in[13];
  const float* w1     = (const float*)d_in[14];
  const float* b1     = (const float*)d_in[15];
  const float* w2     = (const float*)d_in[16];
  const float* b2     = (const float*)d_in[17];
  const float* pw_w   = (const float*)d_in[18];
  const float* pw_b   = (const float*)d_in[19];

  // ---- workspace layout (256B aligned regions) ----
  char* base = (char*)d_ws;
  size_t off = 0;
  auto alloc = [&](size_t bytes) -> char* {
    char* p = base + off;
    off = (off + bytes + 255) & ~(size_t)255;
    return p;
  };
  float* resid = (float*)alloc((size_t)Ntot * Cc * 4);
  float* qf    = (float*)alloc((size_t)Ntot * Cc * 4);
  float* kf    = (float*)alloc((size_t)Ntot * Cc * 4);
  float* vf    = (float*)alloc((size_t)Ntot * Cc * 4);
  u16*   hbf   = (u16*)  alloc((size_t)Ntot * Cc * 2);
  u16*   h1bf  = (u16*)  alloc((size_t)Ntot * FF * 2);
  float* kvbuf = (float*)alloc((size_t)2048 * (DH * DH + DH) * 4);
  const int WQKVO = NB * 2 * Cc * Cc;
  const int WFF   = NB * FF * Cc;
  u16* wq_bf = (u16*)alloc((size_t)WQKVO * 2);
  u16* wk_bf = (u16*)alloc((size_t)WQKVO * 2);
  u16* wv_bf = (u16*)alloc((size_t)WQKVO * 2);
  u16* wo_bf = (u16*)alloc((size_t)WQKVO * 2);
  u16* w1_bf = (u16*)alloc((size_t)WFF * 2);
  u16* w2_bf = (u16*)alloc((size_t)WFF * 2);

  // ---- one-time weight conversion to bf16 ----
  cvt_bf16_kernel<<<(WQKVO + 255) / 256, 256, 0, stream>>>(wq, wq_bf, WQKVO);
  cvt_bf16_kernel<<<(WQKVO + 255) / 256, 256, 0, stream>>>(wk, wk_bf, WQKVO);
  cvt_bf16_kernel<<<(WQKVO + 255) / 256, 256, 0, stream>>>(wv, wv_bf, WQKVO);
  cvt_bf16_kernel<<<(WQKVO + 255) / 256, 256, 0, stream>>>(wo, wo_bf, WQKVO);
  cvt_bf16_kernel<<<(WFF + 255) / 256, 256, 0, stream>>>(w1, w1_bf, WFF);
  cvt_bf16_kernel<<<(WFF + 255) / 256, 256, 0, stream>>>(w2, w2_bf, WFF);

  // ---- embedding + projection + rotary ----
  embed_kernel<<<Ntot, 192, 0, stream>>>(tokens, emb, proj_w, proj_b, resid);

  dim3 g192(Ntot / 32, Cc / 64);  // (1024, 3)
  dim3 g768(Ntot / 32, FF / 64);  // (1024, 12)

  for (int i = 0; i < NB; i++) {
    for (int pass = 0; pass < 2; pass++) {  // 0 = row attn (over L), 1 = col attn (over S)
      int rowMode = (pass == 0);
      size_t lnOff = ((size_t)i * 3 + pass) * Cc;
      ln_kernel<<<Ntot, 192, 0, stream>>>(resid, ln_g + lnOff, ln_b + lnOff, hbf);
      size_t wOff = ((size_t)i * 2 + pass) * Cc * Cc;
      size_t bOff = ((size_t)i * 2 + pass) * Cc;
      gemm_wmma_kernel<Cc><<<g192, 32, 0, stream>>>(hbf, wq_bf + wOff, bq + bOff, qf, nullptr, nullptr, 3);
      gemm_wmma_kernel<Cc><<<g192, 32, 0, stream>>>(hbf, wk_bf + wOff, bk + bOff, kf, nullptr, nullptr, 4);
      gemm_wmma_kernel<Cc><<<g192, 32, 0, stream>>>(hbf, wv_bf + wOff, bv + bOff, vf, nullptr, nullptr, 0);
      int G    = rowMode ? (Bb * Ss) : (Bb * Ll);
      int Ngrp = rowMode ? Ll : Ss;
      kv_reduce_kernel<<<G * Hh, 256, 0, stream>>>(kf, vf, kvbuf, rowMode, Ngrp);
      o_compute_kernel<<<Ntot, 192, 0, stream>>>(qf, kvbuf, hbf, rowMode);
      gemm_wmma_kernel<Cc><<<g192, 32, 0, stream>>>(hbf, wo_bf + wOff, bo + bOff, nullptr, nullptr, resid, 1);
    }
    // FFN
    size_t lnOff = ((size_t)i * 3 + 2) * Cc;
    ln_kernel<<<Ntot, 192, 0, stream>>>(resid, ln_g + lnOff, ln_b + lnOff, hbf);
    gemm_wmma_kernel<Cc><<<g768, 32, 0, stream>>>(hbf, w1_bf + (size_t)i * FF * Cc, b1 + (size_t)i * FF,
                                                  nullptr, h1bf, nullptr, 2);
    gemm_wmma_kernel<FF><<<g192, 32, 0, stream>>>(h1bf, w2_bf + (size_t)i * Cc * FF, b2 + (size_t)i * Cc,
                                                  nullptr, nullptr, resid, 1);
  }

  final_kernel<<<Ntot / 128, 128, 0, stream>>>(resid, pw_w, pw_b, (float*)d_out);
}
